// _MambaBlock_17927193493574
// MI455X (gfx1250) — matrix operand order repaired and checked by reading, not compiled, not run on hardware
//
#include <hip/hip_runtime.h>
#include <hip/hip_bf16.h>

// ---------------- problem constants ----------------
#define NB      4
#define TT      2048
#define CC      1024
#define D_INNER 2048
#define NHEADS  32
#define HEADDIM 64
#define D_STATE 64
#define D_CONVD 2176            // D_INNER + 2*D_STATE
#define D_INPROJ 4256           // 2*D_INNER + 2*D_STATE + NHEADS
#define MTOK    (NB*TT)         // 8192
#define EPSF    1e-5f

// ---------------- WMMA types (CDNA5 wave32) ----------------
typedef __attribute__((ext_vector_type(16))) __bf16 v16bf;
typedef __attribute__((ext_vector_type(8)))  float  v8f;

union Frag { uint4 q[2]; v16bf v; };

__device__ __forceinline__ unsigned short f2bf(float f) {
    unsigned int u = __float_as_uint(f);
    unsigned int r = u + 0x7FFFu + ((u >> 16) & 1u);   // round-to-nearest-even
    return (unsigned short)(r >> 16);
}

__device__ __forceinline__ float wave_sum(float v) {
    #pragma unroll
    for (int o = 16; o > 0; o >>= 1) v += __shfl_xor(v, o, 32);
    return v;
}

// ---------------- 1) LayerNorm + cast to bf16 ----------------
__global__ __launch_bounds__(256) void ln_cast_kernel(
    const float* __restrict__ h, const float* __restrict__ lnw,
    const float* __restrict__ lnb, unsigned short* __restrict__ hn)
{
    const int m = blockIdx.x;
    const int tid = threadIdx.x;
    const float* row = h + (size_t)m * CC;
    float v[4]; float s = 0.f, s2 = 0.f;
    #pragma unroll
    for (int i = 0; i < 4; ++i) {
        v[i] = row[tid + 256 * i];
        s += v[i]; s2 += v[i] * v[i];
    }
    __shared__ float red[16];
    s = wave_sum(s); s2 = wave_sum(s2);
    const int wv = tid >> 5;
    if ((tid & 31) == 0) { red[wv] = s; red[8 + wv] = s2; }
    __syncthreads();
    float ts = 0.f, ts2 = 0.f;
    #pragma unroll
    for (int w = 0; w < 8; ++w) { ts += red[w]; ts2 += red[8 + w]; }
    const float mu  = ts * (1.0f / CC);
    const float var = ts2 * (1.0f / CC) - mu * mu;
    const float inv = rsqrtf(var + EPSF);
    #pragma unroll
    for (int i = 0; i < 4; ++i) {
        const int c = tid + 256 * i;
        const float o = (v[i] - mu) * inv * lnw[c] + lnb[c];
        hn[(size_t)m * CC + c] = f2bf(o);
    }
}

// ---------------- 2) fp32 -> bf16 weight cast ----------------
__global__ void cast_bf16_kernel(const float* __restrict__ src,
                                 unsigned short* __restrict__ dst, int n)
{
    for (int i = blockIdx.x * blockDim.x + threadIdx.x; i < n;
         i += gridDim.x * blockDim.x)
        dst[i] = f2bf(src[i]);
}

// ---------------- 3) WMMA GEMM: out[m,n] = sum_k A[m,k]*W[n,k] (+res) ----
// A: [M,K] bf16 row-major.  W: [Nn,K] bf16 row-major (so B = W^T).
// Block = 256 threads = 8 waves; wave w computes a 32x32 output tile:
// rows m0 = by*256 + w*32, cols n0 = bx*32. 4 accumulators, A frags reused 2x.
__global__ __launch_bounds__(256) void wmma_gemm_nt_kernel(
    const unsigned short* __restrict__ A, const unsigned short* __restrict__ W,
    const float* __restrict__ res, float* __restrict__ out,
    int M, int Nn, int K)
{
    const int tid  = threadIdx.x;
    const int lane = tid & 31;
    const int wave = tid >> 5;
    const int m0 = blockIdx.y * 256 + wave * 32;
    const int n0 = blockIdx.x * 32;

    // A fragment (16x32 bf16): lane<16 -> K offs {0..7,16..23}; lane>=16 -> +8
    const int arow  = m0 + (lane & 15);
    const int akoff = (lane >> 4) << 3;
    // B fragment (32x16 bf16): lane<16 -> K offs {0..7,16..23} of col n; lane>=16 -> +8
    const int bcol  = n0 + (lane & 15);
    const int bkoff = (lane >> 4) << 3;

    const unsigned short* aptr0 = A + (size_t)arow * K + akoff;
    const unsigned short* aptr1 = aptr0 + (size_t)16 * K;
    const unsigned short* bptr0 = W + (size_t)bcol * K + bkoff;
    const unsigned short* bptr1 = bptr0 + (size_t)16 * K;

    v8f acc00 = {}; v8f acc01 = {}; v8f acc10 = {}; v8f acc11 = {};
    for (int k0 = 0; k0 < K; k0 += 32) {
        Frag a0, a1, b0, b1;
        a0.q[0] = *(const uint4*)(aptr0 + k0);
        a0.q[1] = *(const uint4*)(aptr0 + k0 + 16);
        a1.q[0] = *(const uint4*)(aptr1 + k0);
        a1.q[1] = *(const uint4*)(aptr1 + k0 + 16);
        b0.q[0] = *(const uint4*)(bptr0 + k0);
        b0.q[1] = *(const uint4*)(bptr0 + k0 + 16);
        b1.q[0] = *(const uint4*)(bptr1 + k0);
        b1.q[1] = *(const uint4*)(bptr1 + k0 + 16);
        // CDNA5 prefetch (global_prefetch_b8) for the next K tile
        __builtin_prefetch(aptr0 + k0 + 32, 0, 3);
        __builtin_prefetch(aptr1 + k0 + 32, 0, 3);
        __builtin_prefetch(bptr0 + k0 + 32, 0, 3);
        __builtin_prefetch(bptr1 + k0 + 32, 0, 3);
        acc00 = __builtin_amdgcn_wmma_f32_16x16x32_bf16(
                    false, a0.v, false, b0.v, (short)0, acc00, false, false);
        acc01 = __builtin_amdgcn_wmma_f32_16x16x32_bf16(
                    false, a0.v, false, b1.v, (short)0, acc01, false, false);
        acc10 = __builtin_amdgcn_wmma_f32_16x16x32_bf16(
                    false, a1.v, false, b0.v, (short)0, acc10, false, false);
        acc11 = __builtin_amdgcn_wmma_f32_16x16x32_bf16(
                    false, a1.v, false, b1.v, (short)0, acc11, false, false);
    }
    // C/D layout: lane<16 -> N=lane, M=r ; lane>=16 -> N=lane-16, M=8+r
    const int mrow = m0 + ((lane >> 4) << 3);
    const int ncol = n0 + (lane & 15);
    #pragma unroll
    for (int r = 0; r < 8; ++r) {
        const size_t i0 = (size_t)(mrow + r) * Nn + ncol;        // rows m0..m0+15
        const size_t i1 = (size_t)(mrow + 16 + r) * Nn + ncol;   // rows m0+16..+31
        float o00 = acc00[r], o01 = acc01[r], o10 = acc10[r], o11 = acc11[r];
        if (res) {
            o00 += res[i0]; o01 += res[i0 + 16];
            o10 += res[i1]; o11 += res[i1 + 16];
        }
        out[i0] = o00; out[i0 + 16] = o01;
        out[i1] = o10; out[i1 + 16] = o11;
    }
}

// ---------------- 4) depthwise conv (K=4, causal) + SiLU ----------------
__global__ __launch_bounds__(128) void conv_silu_kernel(
    const float* __restrict__ zx, const float* __restrict__ cw,
    const float* __restrict__ cb, float* __restrict__ xc)
{
    const int m = blockIdx.x;                       // n*T + t
    const int c = blockIdx.y * 128 + threadIdx.x;   // 0..2175
    const int t = m & (TT - 1);
    const float w0 = cw[c * 4 + 0], w1 = cw[c * 4 + 1];
    const float w2 = cw[c * 4 + 2], w3 = cw[c * 4 + 3];
    const size_t base = (size_t)m * D_INPROJ + D_INNER + c;
    float acc = cb[c] + zx[base] * w3;
    if (t >= 1) acc += zx[base -     D_INPROJ] * w2;
    if (t >= 2) acc += zx[base - 2 * D_INPROJ] * w1;
    if (t >= 3) acc += zx[base - 3 * D_INPROJ] * w0;
    const float sg = 1.f / (1.f + __expf(-acc));
    xc[(size_t)m * D_CONVD + c] = acc * sg;
}

// ---------------- 5) dt = softplus(raw + bias), dA = exp(dt*A) -----------
__global__ __launch_bounds__(32) void dt_kernel(
    const float* __restrict__ zx, const float* __restrict__ dt_bias,
    const float* __restrict__ A_log, float* __restrict__ dtb,
    float* __restrict__ dab)
{
    const int m = blockIdx.x;
    const int hh = threadIdx.x;   // 0..31
    const float raw = zx[(size_t)m * D_INPROJ + (D_INNER + D_CONVD) + hh] + dt_bias[hh];
    const float dt = (raw > 20.f) ? raw : log1pf(__expf(raw));
    const float Ah = -__expf(A_log[hh]);
    dtb[m * NHEADS + hh] = dt;
    dab[m * NHEADS + hh] = __expf(dt * Ah);
}

// ---------------- 6) sequential SSM scan ----------------
// One block per (batch, head); 64 lanes, lane p owns state[p, 0..63] in VGPRs.
// B/C vectors staged memory->LDS via CDNA5 async-LDS loads (ASYNCcnt tracked).
__global__ __launch_bounds__(64) void scan_kernel(
    const float* __restrict__ xc, const float* __restrict__ dtb,
    const float* __restrict__ dab, const float* __restrict__ Dvec,
    float* __restrict__ yb)
{
    const int n = blockIdx.x >> 5;
    const int hh = blockIdx.x & 31;
    const int p = threadIdx.x;
    __shared__ float sB[D_STATE], sC[D_STATE];
    float st[D_STATE];
    #pragma unroll
    for (int s = 0; s < D_STATE; ++s) st[s] = 0.f;
    const float Dh = Dvec[hh];
    // LDS byte offsets for this lane's slot (generic shared addr low 32 bits
    // equal the LDS offset per the gfx1250 aperture mapping).
    const unsigned ldsB = (unsigned)(size_t)&sB[p];
    const unsigned ldsC = (unsigned)(size_t)&sC[p];
    for (int t = 0; t < TT; ++t) {
        const size_t mrow  = (size_t)(n * TT + t);
        const size_t cbase = mrow * D_CONVD;
        const unsigned long long gB =
            (unsigned long long)(size_t)(xc + cbase + D_INNER + p);
        const unsigned long long gC = gB + (unsigned long long)D_STATE * 4u;
        // GLOBAL_LOAD_ASYNC_TO_LDS_B32: each lane copies one dword from global memory into LDS
        asm volatile("global_load_async_to_lds_b32 %0, %1, off"
                     :: "v"(ldsB), "v"(gB) : "memory");
        asm volatile("global_load_async_to_lds_b32 %0, %1, off"
                     :: "v"(ldsC), "v"(gC) : "memory");
        asm volatile("s_wait_asynccnt 0" ::: "memory");
        __syncthreads();
        const float xv   = xc[cbase + hh * HEADDIM + p];
        const float dav  = dab[mrow * NHEADS + hh];
        const float coef = dtb[mrow * NHEADS + hh] * xv;
        float y = 0.f;
        #pragma unroll
        for (int s = 0; s < D_STATE; ++s) {
            st[s] = st[s] * dav + coef * sB[s];
            y += st[s] * sC[s];
        }
        yb[mrow * D_INNER + hh * HEADDIM + p] = y + Dh * xv;
        __syncthreads();
    }
}

// ---------------- 7) gate * silu(z), RMSNorm, cast bf16 ----------------
__global__ __launch_bounds__(256) void gate_rms_kernel(
    const float* __restrict__ yb, const float* __restrict__ zx,
    const float* __restrict__ rms_w, unsigned short* __restrict__ yn)
{
    const int m = blockIdx.x;
    const int tid = threadIdx.x;
    float yv[8]; float ss = 0.f;
    #pragma unroll
    for (int i = 0; i < 8; ++i) {
        const int d = tid + 256 * i;
        const float z = zx[(size_t)m * D_INPROJ + d];
        const float y = yb[(size_t)m * D_INNER + d] * (z / (1.f + __expf(-z)));
        yv[i] = y; ss += y * y;
    }
    __shared__ float red[8];
    ss = wave_sum(ss);
    if ((tid & 31) == 0) red[tid >> 5] = ss;
    __syncthreads();
    float ts = 0.f;
    #pragma unroll
    for (int w = 0; w < 8; ++w) ts += red[w];
    const float inv = rsqrtf(ts * (1.0f / D_INNER) + EPSF);
    #pragma unroll
    for (int i = 0; i < 8; ++i) {
        const int d = tid + 256 * i;
        yn[(size_t)m * D_INNER + d] = f2bf(yv[i] * inv * rms_w[d]);
    }
}

// ---------------- launch ----------------
extern "C" void kernel_launch(void* const* d_in, const int* in_sizes, int n_in,
                              void* d_out, int out_size, void* d_ws, size_t ws_size,
                              hipStream_t stream)
{
    const float* h       = (const float*)d_in[0];
    const float* ln_w    = (const float*)d_in[1];
    const float* ln_b    = (const float*)d_in[2];
    const float* in_w    = (const float*)d_in[3];
    const float* conv_w  = (const float*)d_in[4];
    const float* conv_b  = (const float*)d_in[5];
    const float* dt_bias = (const float*)d_in[6];
    const float* A_log   = (const float*)d_in[7];
    const float* Dvec    = (const float*)d_in[8];
    const float* rms_w   = (const float*)d_in[9];
    const float* out_w   = (const float*)d_in[10];
    float* out = (float*)d_out;

    char* p = (char*)d_ws;
    auto take = [&](size_t bytes) -> char* {
        char* r = p; p += (bytes + 255) & ~(size_t)255; return r;
    };
    unsigned short* hn   = (unsigned short*)take((size_t)MTOK * CC * 2);
    unsigned short* win  = (unsigned short*)take((size_t)D_INPROJ * CC * 2);
    unsigned short* wout = (unsigned short*)take((size_t)CC * D_INNER * 2);
    float* zx  = (float*)take((size_t)MTOK * D_INPROJ * 4);
    float* xc  = (float*)take((size_t)MTOK * D_CONVD * 4);
    float* dtb = (float*)take((size_t)MTOK * NHEADS * 4);
    float* dab = (float*)take((size_t)MTOK * NHEADS * 4);
    float* yb  = (float*)take((size_t)MTOK * D_INNER * 4);
    unsigned short* yn = (unsigned short*)take((size_t)MTOK * D_INNER * 2);

    // 1) LN + bf16 cast of activations
    ln_cast_kernel<<<MTOK, 256, 0, stream>>>(h, ln_w, ln_b, hn);
    // 2) bf16 weight casts
    cast_bf16_kernel<<<2048, 256, 0, stream>>>(in_w, win, D_INPROJ * CC);
    cast_bf16_kernel<<<1024, 256, 0, stream>>>(out_w, wout, CC * D_INNER);
    // 3) in_proj GEMM: zx[8192,4256] = hn * win^T   (bf16 WMMA, fp32 acc)
    wmma_gemm_nt_kernel<<<dim3(D_INPROJ / 32, MTOK / 256), 256, 0, stream>>>(
        hn, win, nullptr, zx, MTOK, D_INPROJ, CC);
    // 4) causal conv4 + SiLU over xBC channels
    conv_silu_kernel<<<dim3(MTOK, D_CONVD / 128), 128, 0, stream>>>(
        zx, conv_w, conv_b, xc);
    // 5) dt / dA
    dt_kernel<<<MTOK, 32, 0, stream>>>(zx, dt_bias, A_log, dtb, dab);
    // 6) sequential scan (128 independent (batch,head) recurrences)
    scan_kernel<<<NB * NHEADS, 64, 0, stream>>>(xc, dtb, dab, Dvec, yb);
    // 7) gate + RMSNorm + bf16 cast
    gate_rms_kernel<<<MTOK, 256, 0, stream>>>(yb, zx, rms_w, yn);
    // 8) out_proj GEMM fused with residual: out = h + yn * wout^T
    wmma_gemm_nt_kernel<<<dim3(CC / 32, MTOK / 256), 256, 0, stream>>>(
        yn, wout, h, out, MTOK, CC, D_INNER);
    (void)in_sizes; (void)n_in; (void)out_size; (void)ws_size;
}
